// MultiHeadAttention_26276609917518
// MI455X (gfx1250) — matrix operand order repaired and checked by reading, not compiled, not run on hardware
//
#include <hip/hip_runtime.h>
#include <hip/hip_bf16.h>
#include <stdint.h>

// ---------------------------------------------------------------------------
// MHA for MI455X (gfx1250): bf16 WMMA pipeline, double-buffered async
// global->LDS staging (ASYNCcnt ping-pong).
//   1) convert weights f32 -> bf16
//   2) qkv = x @ w_qkv^T  (WMMA; x f32->bf16 via VGPRs, W async->LDS 2-deep),
//      scatter to q[N,hd], k[N,hd], v^T[hd,N] per (b,h) in bf16
//   3) flash attention per (b,h): K/V async->LDS 2-deep, online softmax, WMMA
//   4) out = attn @ w_proj^T (WMMA, both operands async->LDS 2-deep) -> f32
// ---------------------------------------------------------------------------

#define DIMD   2048
#define NHEAD  16
#define HD     128
#define BATCH  8
#define SEQ    4096
#define ROWS   (BATCH * SEQ)   // 32768

typedef __attribute__((ext_vector_type(16))) __bf16 bf16x16;
typedef __attribute__((ext_vector_type(8)))  __bf16 bf16x8;
typedef __attribute__((ext_vector_type(8)))  float  f32x8;
typedef __attribute__((address_space(3))) char* lds_cp;

__device__ __forceinline__ uint16_t bf16_bits(float f) {
  union { float f; uint32_t u; } a; a.f = f;
  uint32_t r = a.u + 0x7FFFu + ((a.u >> 16) & 1u);   // round-to-nearest-even
  return (uint16_t)(r >> 16);
}
__device__ __forceinline__ __bf16 to_bf16(float f) {
  uint16_t h = bf16_bits(f);
  __bf16 b; __builtin_memcpy(&b, &h, 2);
  return b;
}
__device__ __forceinline__ f32x8 zero8() {
  f32x8 z = {0.f,0.f,0.f,0.f,0.f,0.f,0.f,0.f};
  return z;
}
__device__ __forceinline__ f32x8 wmma_bf16(bf16x16 a, bf16x16 b, f32x8 c) {
  // D = A(16x32) * B(32x16) + C(16x16 f32)
  return __builtin_amdgcn_wmma_f32_16x16x32_bf16(false, a, false, b, (short)0, c,
                                                 false, false);
}
__device__ __forceinline__ bf16x16 combine8(bf16x8 lo, bf16x8 hi) {
  return __builtin_shufflevector(lo, hi, 0,1,2,3,4,5,6,7,8,9,10,11,12,13,14,15);
}

// Async 16-byte global -> LDS copy (per active lane). Tracked by ASYNCcnt.
__device__ __forceinline__ void async_ld_b128(void* lds_ptr, const void* gptr) {
  uint32_t loff = (uint32_t)(uintptr_t)(lds_cp)lds_ptr;   // raw LDS byte offset
  asm volatile("global_load_async_to_lds_b128 %0, %1, off"
               :: "v"(loff), "v"(gptr) : "memory");
}
// Async loads complete in order (per wave): waiting ASYNCcnt <= N guarantees
// all but the newest N async ops have landed.
__device__ __forceinline__ void wait_async0() {
  asm volatile("s_wait_asynccnt 0x0" ::: "memory");
}
__device__ __forceinline__ void wait_async1() {
  asm volatile("s_wait_asynccnt 0x1" ::: "memory");
}
__device__ __forceinline__ void wait_async3() {
  asm volatile("s_wait_asynccnt 0x3" ::: "memory");
}
__device__ __forceinline__ void wait_async4() {
  asm volatile("s_wait_asynccnt 0x4" ::: "memory");
}

// A fragment (16x32, bf16) from LDS row-major [..][LDAP]; rows mbase..+15.
// ISA: lane m = lane%16; elems 0..7 -> K = 8*laneHi.., elems 8..15 -> K=16+8*laneHi..
template<int LDAP>
__device__ __forceinline__ bf16x16 load_a_frag(const __bf16* lds, int mbase, int kbase,
                                               int lane16, int laneHi) {
  const __bf16* p = lds + (size_t)(mbase + lane16) * LDAP + kbase;
  bf16x8 lo = *(const bf16x8*)(p + 8 * laneHi);
  bf16x8 hi = *(const bf16x8*)(p + 16 + 8 * laneHi);
  return combine8(lo, hi);
}

// B fragment (32x16, bf16): lds holds B^T row-major: lds[n][k].
// ISA: lane n = lane%16; elems 0..7 -> K = 8*laneHi.., elems 8..15 -> K=16+8*laneHi..
template<int LDBP>
__device__ __forceinline__ bf16x16 load_b_frag(const __bf16* lds, int nbase, int kbase,
                                               int lane16, int laneHi) {
  const __bf16* p = lds + (size_t)(nbase + lane16) * LDBP + kbase;
  bf16x8 lo = *(const bf16x8*)(p + 8 * laneHi);
  bf16x8 hi = *(const bf16x8*)(p + 16 + 8 * laneHi);
  return combine8(lo, hi);
}

// ---------------------------------------------------------------------------
__global__ void convert_f32_bf16(const float* __restrict__ in,
                                 __bf16* __restrict__ out, int nelem) {
  int i = blockIdx.x * blockDim.x + threadIdx.x;
  int stride = gridDim.x * blockDim.x;
  for (; i < nelem; i += stride) out[i] = to_bf16(in[i]);
}

// ---------------------------------------------------------------------------
// QKV GEMM: C[32768 x 6144] = x[32768 x 2048](f32) @ w_qkv[6144 x 2048]^T(bf16)
// WG tile 128x64, 8 waves of 32x32. B tile async->LDS double-buffered
// (1 async op/wave/batch: lanes 0-15 of every wave). A tile f32->bf16 in VGPRs.
__global__ __launch_bounds__(256)
void qkv_gemm_kernel(const float* __restrict__ x, const __bf16* __restrict__ wq,
                     __bf16* __restrict__ qws, __bf16* __restrict__ kws,
                     __bf16* __restrict__ vws) {
  __shared__ __bf16 ldsA[128][40];
  __shared__ __bf16 ldsB[2][64][40];
  const int tid = threadIdx.x;
  const int wid = tid >> 5;
  const int lane = tid & 31;
  const int lane16 = lane & 15;
  const int laneHi = lane >> 4;
  const int rowBase = blockIdx.y * 128;
  const int colBase = blockIdx.x * 64;
  const int mw = (wid & 3) * 32;
  const int nw = (wid >> 2) * 32;

  f32x8 acc[2][2];
  acc[0][0] = zero8(); acc[0][1] = zero8();
  acc[1][0] = zero8(); acc[1][1] = zero8();

  // stage B tile for step k0 into buffer buf (128 x b128, 1 per wave: lanes 0-15)
  auto stageB = [&](int k0, int buf) {
    if (laneHi == 0) {
      int idx = wid * 16 + lane16;      // 0..127
      int r = idx >> 1;
      int c8 = (idx & 1) * 8;
      async_ld_b128(&ldsB[buf][r][c8],
                    wq + (size_t)(colBase + r) * DIMD + k0 + c8);
    }
  };

  stageB(0, 0);
  for (int k0 = 0, it = 0; k0 < DIMD; k0 += 32, ++it) {
    const int cur = it & 1;
    const bool more = (k0 + 32 < DIMD);
    if (more) stageB(k0 + 32, cur ^ 1);

    // Stage A: 128 rows x 32 f32 -> bf16 LDS (conversion -> VGPR path)
#pragma unroll
    for (int i = 0; i < 4; ++i) {
      int idx = tid + i * 256;            // 0..1023 float4s
      int r = idx >> 3;
      int kq = (idx & 7) * 4;
      float4 v = *(const float4*)(x + (size_t)(rowBase + r) * DIMD + k0 + kq);
      ushort4 hv;
      hv.x = bf16_bits(v.x); hv.y = bf16_bits(v.y);
      hv.z = bf16_bits(v.z); hv.w = bf16_bits(v.w);
      *(ushort4*)((uint16_t*)&ldsA[r][kq]) = hv;
    }
    // L2 prefetch of next A tile
    if (more) {
      int r  = tid >> 1;
      int cc = (tid & 1) * 16;
      __builtin_prefetch(x + (size_t)(rowBase + r) * DIMD + k0 + 32 + cc, 0, 1);
    }
    if (more) wait_async1(); else wait_async0();   // batch for k0 has landed
    __syncthreads();

    bf16x16 af[2], bfr[2];
    af[0]  = load_a_frag<40>(&ldsA[0][0],      mw,      0, lane16, laneHi);
    af[1]  = load_a_frag<40>(&ldsA[0][0],      mw + 16, 0, lane16, laneHi);
    bfr[0] = load_b_frag<40>(&ldsB[cur][0][0], nw,      0, lane16, laneHi);
    bfr[1] = load_b_frag<40>(&ldsB[cur][0][0], nw + 16, 0, lane16, laneHi);
#pragma unroll
    for (int mi = 0; mi < 2; ++mi)
#pragma unroll
      for (int ni = 0; ni < 2; ++ni)
        acc[mi][ni] = wmma_bf16(af[mi], bfr[ni], acc[mi][ni]);
    __syncthreads();   // all reads done before buffers are re-staged
  }

  // Epilogue: scatter to q[N,hd] / k[N,hd] / v^T[hd,N] per (b,h) in bf16.
#pragma unroll
  for (int mi = 0; mi < 2; ++mi) {
#pragma unroll
    for (int ni = 0; ni < 2; ++ni) {
      int col = colBase + nw + ni * 16 + lane16;   // 0..6143
      int t   = col >> 11;                         // 0=q 1=k 2=v
      int rem = col & 2047;
      int h   = rem >> 7;
      int d   = rem & 127;
#pragma unroll
      for (int r = 0; r < 8; ++r) {
        int row = rowBase + mw + mi * 16 + r + 8 * laneHi;
        int b = row >> 12;
        int n = row & 4095;
        __bf16 val = to_bf16(acc[mi][ni][r]);
        size_t bh = (size_t)(b * NHEAD + h);
        if (t == 0)      qws[(bh * SEQ + n) * HD + d] = val;
        else if (t == 1) kws[(bh * SEQ + n) * HD + d] = val;
        else             vws[bh * (size_t)SEQ * HD + (size_t)d * SEQ + n] = val;
      }
    }
  }
}

// ---------------------------------------------------------------------------
// Flash attention: one wave = 16 query rows; WG of 8 waves = 128 rows.
// Key blocks of 32; K and V^T async->LDS, double-buffered (4 ops/wave/batch).
__global__ __launch_bounds__(256)
void flash_attn_kernel(const __bf16* __restrict__ qws, const __bf16* __restrict__ kws,
                       const __bf16* __restrict__ vws, __bf16* __restrict__ ows) {
  __shared__ __bf16 ldsK[2][32][136];   // [buf][j_local][d]
  __shared__ __bf16 ldsV[2][128][40];   // [buf][d][j_local]   (V^T)
  __shared__ __bf16 ldsP[8][16][40];    // per-wave P transpose buffer

  const int tid = threadIdx.x;
  const int wid = tid >> 5;
  const int lane = tid & 31;
  const int lane16 = lane & 15;
  const int laneHi = lane >> 4;
  const int bh = blockIdx.y;                 // 0..127
  const int b = bh >> 4, h = bh & 15;
  const int qbase = blockIdx.x * 128 + wid * 16;

  const size_t head = (size_t)bh * SEQ * HD;
  const __bf16* Q = qws + head;
  const __bf16* K = kws + head;
  const __bf16* V = vws + head;              // hd x SEQ

  // Preload Q fragments (4 k-chunks of 32 over hd=128)
  bf16x16 qf[4];
#pragma unroll
  for (int kc = 0; kc < 4; ++kc) {
    const __bf16* p = Q + (size_t)(qbase + lane16) * HD + kc * 32;
    bf16x8 lo = *(const bf16x8*)(p + 8 * laneHi);
    bf16x8 hi = *(const bf16x8*)(p + 16 + 8 * laneHi);
    qf[kc] = combine8(lo, hi);
  }

  f32x8 o[8];
#pragma unroll
  for (int nt = 0; nt < 8; ++nt) o[nt] = zero8();
  float m[8], lsum[8];
#pragma unroll
  for (int r = 0; r < 8; ++r) { m[r] = -1.0e30f; lsum[r] = 0.f; }
  const float scale = 0.08838834764831845f;  // 1/sqrt(128)

  auto stageKV = [&](int jb, int buf) {
#pragma unroll
    for (int i = 0; i < 2; ++i) {            // K: 32 x 128 bf16 = 512 x b128
      int idx = tid + i * 256;
      int r = idx >> 4;
      int c8 = (idx & 15) * 8;
      async_ld_b128(&ldsK[buf][r][c8], K + (size_t)(jb * 32 + r) * HD + c8);
    }
#pragma unroll
    for (int i = 0; i < 2; ++i) {            // V^T: 128 x 32 bf16 = 512 x b128
      int idx = tid + i * 256;
      int r = idx >> 2;
      int c8 = (idx & 3) * 8;
      async_ld_b128(&ldsV[buf][r][c8], V + (size_t)r * SEQ + jb * 32 + c8);
    }
  };

  stageKV(0, 0);
  for (int jb = 0; jb < SEQ / 32; ++jb) {
    const int cur = jb & 1;
    const bool more = (jb + 1 < SEQ / 32);
    if (more) { stageKV(jb + 1, cur ^ 1); wait_async4(); }
    else      { wait_async0(); }
    __syncthreads();

    // S (16x32) = Q * K^T
    f32x8 s0 = zero8(), s1 = zero8();
#pragma unroll
    for (int kc = 0; kc < 4; ++kc) {
      bf16x16 b0 = load_b_frag<136>(&ldsK[cur][0][0], 0,  kc * 32, lane16, laneHi);
      bf16x16 b1 = load_b_frag<136>(&ldsK[cur][0][0], 16, kc * 32, lane16, laneHi);
      s0 = wmma_bf16(qf[kc], b0, s0);
      s1 = wmma_bf16(qf[kc], b1, s1);
    }

    // Online softmax update, write P into per-wave LDS (A-layout source)
#pragma unroll
    for (int r = 0; r < 8; ++r) {
      float a0 = s0[r] * scale;
      float a1 = s1[r] * scale;
      float v = fmaxf(a0, a1);
      v = fmaxf(v, __shfl_xor(v, 1));
      v = fmaxf(v, __shfl_xor(v, 2));
      v = fmaxf(v, __shfl_xor(v, 4));
      v = fmaxf(v, __shfl_xor(v, 8));
      float mnew = fmaxf(m[r], v);
      float corr = __expf(m[r] - mnew);
      m[r] = mnew;
      float p0 = __expf(a0 - mnew);
      float p1 = __expf(a1 - mnew);
      lsum[r] = lsum[r] * corr + p0 + p1;
#pragma unroll
      for (int nt = 0; nt < 8; ++nt) o[nt][r] *= corr;
      int prow = r + 8 * laneHi;               // C-layout row of this lane
      ldsP[wid][prow][lane16]      = to_bf16(p0);
      ldsP[wid][prow][16 + lane16] = to_bf16(p1);
    }
    asm volatile("s_wait_dscnt 0x0" ::: "memory");  // wave-local LDS RAW fence

    // P fragment (16x32 A-layout)
    bf16x16 pf;
    {
      const __bf16* p = &ldsP[wid][lane16][0];
      bf16x8 lo = *(const bf16x8*)(p + 8 * laneHi);
      bf16x8 hi = *(const bf16x8*)(p + 16 + 8 * laneHi);
      pf = combine8(lo, hi);
    }
    // O += P * V  (8 n-tiles over hd=128)
#pragma unroll
    for (int nt = 0; nt < 8; ++nt) {
      bf16x16 vf = combine8(*(const bf16x8*)&ldsV[cur][nt * 16 + lane16][8 * laneHi],
                            *(const bf16x8*)&ldsV[cur][nt * 16 + lane16][16 + 8 * laneHi]);
      o[nt] = wmma_bf16(pf, vf, o[nt]);
    }
    __syncthreads();   // all reads done before buffer is re-staged
  }

  // Finalize: reduce row sums across the 16-lane half, normalize, store bf16
#pragma unroll
  for (int r = 0; r < 8; ++r) {
    float v = lsum[r];
    v += __shfl_xor(v, 1);
    v += __shfl_xor(v, 2);
    v += __shfl_xor(v, 4);
    v += __shfl_xor(v, 8);
    lsum[r] = 1.0f / v;
  }
  const int bcol = h * HD;
#pragma unroll
  for (int nt = 0; nt < 8; ++nt) {
#pragma unroll
    for (int r = 0; r < 8; ++r) {
      int n = qbase + r + 8 * laneHi;
      size_t off = (size_t)(b * SEQ + n) * DIMD + bcol + nt * 16 + lane16;
      ows[off] = to_bf16(o[nt][r] * lsum[r]);
    }
  }
}

// ---------------------------------------------------------------------------
// Proj GEMM: out[32768 x 2048](f32) = attn[32768 x 2048](bf16) @ w_proj^T(bf16)
// Both operand tiles async->LDS, double-buffered (3 ops/wave/batch).
__global__ __launch_bounds__(256)
void proj_gemm_kernel(const __bf16* __restrict__ a, const __bf16* __restrict__ wp,
                      float* __restrict__ out) {
  __shared__ __bf16 ldsA[2][128][40];
  __shared__ __bf16 ldsB[2][64][40];
  const int tid = threadIdx.x;
  const int wid = tid >> 5;
  const int lane = tid & 31;
  const int lane16 = lane & 15;
  const int laneHi = lane >> 4;
  const int rowBase = blockIdx.y * 128;
  const int colBase = blockIdx.x * 64;
  const int mw = (wid & 3) * 32;
  const int nw = (wid >> 2) * 32;

  f32x8 acc[2][2];
  acc[0][0] = zero8(); acc[0][1] = zero8();
  acc[1][0] = zero8(); acc[1][1] = zero8();

  auto stageAB = [&](int k0, int buf) {
#pragma unroll
    for (int i = 0; i < 2; ++i) {            // A: 128 x 32 bf16 = 512 x b128
      int idx = tid + i * 256;
      int r = idx >> 2;
      int c8 = (idx & 3) * 8;
      async_ld_b128(&ldsA[buf][r][c8],
                    a + (size_t)(rowBase + r) * DIMD + k0 + c8);
    }
    if (laneHi == 0) {                        // B: 128 x b128, lanes 0-15/wave
      int idx = wid * 16 + lane16;
      int r = idx >> 1;
      int c8 = (idx & 1) * 8;
      async_ld_b128(&ldsB[buf][r][c8],
                    wp + (size_t)(colBase + r) * DIMD + k0 + c8);
    }
  };

  stageAB(0, 0);
  for (int k0 = 0, it = 0; k0 < DIMD; k0 += 32, ++it) {
    const int cur = it & 1;
    const bool more = (k0 + 32 < DIMD);
    if (more) { stageAB(k0 + 32, cur ^ 1); wait_async3(); }
    else      { wait_async0(); }
    __syncthreads();

    bf16x16 af[2], bfr[2];
    af[0]  = load_a_frag<40>(&ldsA[cur][0][0], mw,      0, lane16, laneHi);
    af[1]  = load_a_frag<40>(&ldsA[cur][0][0], mw + 16, 0, lane16, laneHi);
    bfr[0] = load_b_frag<40>(&ldsB[cur][0][0], nw,      0, lane16, laneHi);
    bfr[1] = load_b_frag<40>(&ldsB[cur][0][0], nw + 16, 0, lane16, laneHi);
#pragma unroll
    for (int mi = 0; mi < 2; ++mi)
#pragma unroll
      for (int ni = 0; ni < 2; ++ni)
        acc[mi][ni] = wmma_bf16(af[mi], bfr[ni], acc[mi][ni]);
    __syncthreads();   // all reads done before buffers are re-staged
  }

#pragma unroll
  for (int mi = 0; mi < 2; ++mi) {
#pragma unroll
    for (int ni = 0; ni < 2; ++ni) {
      int col = colBase + nw + ni * 16 + lane16;
#pragma unroll
      for (int r = 0; r < 8; ++r) {
        int row = rowBase + mw + mi * 16 + r + 8 * laneHi;
        out[(size_t)row * DIMD + col] = acc[mi][ni][r];
      }
    }
  }
}

// ---------------------------------------------------------------------------
extern "C" void kernel_launch(void* const* d_in, const int* in_sizes, int n_in,
                              void* d_out, int out_size, void* d_ws, size_t ws_size,
                              hipStream_t stream) {
  (void)in_sizes; (void)n_in; (void)out_size; (void)ws_size;
  const float* x      = (const float*)d_in[0];
  const float* w_qkv  = (const float*)d_in[1];
  const float* w_proj = (const float*)d_in[2];
  float* out = (float*)d_out;

  char* ws = (char*)d_ws;
  size_t off = 0;
  __bf16* wq_bf = (__bf16*)(ws + off); off += (size_t)3 * DIMD * DIMD * 2;
  __bf16* wp_bf = (__bf16*)(ws + off); off += (size_t)DIMD * DIMD * 2;
  const size_t qkv_elems = (size_t)ROWS * DIMD;            // 67,108,864
  __bf16* qws = (__bf16*)(ws + off); off += qkv_elems * 2;
  __bf16* kws = (__bf16*)(ws + off); off += qkv_elems * 2;
  __bf16* vws = (__bf16*)(ws + off); off += qkv_elems * 2;
  __bf16* aws = (__bf16*)(ws + off); off += qkv_elems * 2;

  convert_f32_bf16<<<2048, 256, 0, stream>>>(w_qkv, wq_bf, 3 * DIMD * DIMD);
  convert_f32_bf16<<<1024, 256, 0, stream>>>(w_proj, wp_bf, DIMD * DIMD);

  qkv_gemm_kernel<<<dim3(6144 / 64, ROWS / 128), 256, 0, stream>>>(
      x, wq_bf, qws, kws, vws);

  flash_attn_kernel<<<dim3(SEQ / 128, BATCH * NHEAD), 256, 0, stream>>>(
      qws, kws, vws, aws);

  proj_gemm_kernel<<<dim3(DIMD / 64, ROWS / 128), 256, 0, stream>>>(
      aws, wp_bf, out);
}
